// GAT_48335561949801
// MI455X (gfx1250) — matrix operand order repaired and checked by reading, not compiled, not run on hardware
//
#include <hip/hip_runtime.h>
#include <math.h>

// ---------------------------------------------------------------- constants
#define N_NODES 20000
#define N_EDGES 256000
#define F_IN    1024
#define HID     1000
#define N_CLS   1000
#define LDP     1024          // padded feature stride (elements)
#define MTILES  (N_NODES/16)  // 1250
#define NEG_SLOPE 0.2f

typedef _Float16 h8  __attribute__((ext_vector_type(8)));
typedef _Float16 h16 __attribute__((ext_vector_type(16)));
typedef float    f8  __attribute__((ext_vector_type(8)));

// ------------------------------------------------ float <-> ordered-uint (for atomic max on floats)
__device__ __forceinline__ unsigned f2ord(float f) {
    unsigned u = __float_as_uint(f);
    return (u & 0x80000000u) ? ~u : (u | 0x80000000u);
}
__device__ __forceinline__ float ord2f(unsigned u) {
    return (u & 0x80000000u) ? __uint_as_float(u & 0x7fffffffu) : __uint_as_float(~u);
}

// ---------------------------------------------------------------- converts
__global__ __launch_bounds__(256)
void k_cvt_x(const float* __restrict__ x, _Float16* __restrict__ xf, int n) {
    int i = blockIdx.x * 256 + threadIdx.x;
    if (i < n) xf[i] = (_Float16)x[i];
}

// weights [rows, K] f32 -> [1024, 1024] f16, zero-padded
__global__ __launch_bounds__(256)
void k_cvt_w(const float* __restrict__ w, _Float16* __restrict__ wf, int rows, int K) {
    int i = blockIdx.x * 256 + threadIdx.x;          // 1024*1024 ids
    int r = i >> 10, k = i & (LDP - 1);
    float v = (r < rows && k < K) ? w[(size_t)r * K + k] : 0.f;
    wf[i] = (_Float16)v;
}

// ---------------------------------------------------------------- WMMA GEMM
// out[m, c] = sum_k A[m,k] * W[c,k] + bias[c]
// A: [N_NODES, LDP] f16 (row major), W: [1024, LDP] f16 (row major, zero padded)
// out: [N_NODES, LDP] f32 (cols >= C left untouched)
// One wave -> 16x64 output (4 n-tiles, A fragment reused 4x). 8 waves/block.
__global__ __launch_bounds__(256)
void k_gemm_bias(const _Float16* __restrict__ A, const _Float16* __restrict__ W,
                 const float* __restrict__ bias, float* __restrict__ out, int C) {
    const int wid   = threadIdx.x >> 5;
    const int lane  = threadIdx.x & 31;
    const int mtile = blockIdx.x * 8 + wid;
    if (mtile >= MTILES) return;                 // wave-uniform exit
    const int row0 = mtile << 4;
    const int col0 = blockIdx.y << 6;
    const int lr   = lane & 15;
    const int hi   = (lane & 16) ? 1 : 0;

    const _Float16* arow = A + (size_t)(row0 + lr) * LDP + (hi ? 16 : 0);
    f8 acc[4] = {};

    for (int kk = 0; kk < LDP; kk += 32) {
        // A fragment: 16x32 f16: lane<16 -> K {0..15}, lane>=16 -> K {16..31}
        const _Float16* ap = arow + kk;
        h8 alo = *(const h8*)(ap);
        h8 ahi = *(const h8*)(ap + 8);
        h16 a  = __builtin_shufflevector(alo, ahi,
                    0,1,2,3,4,5,6,7,8,9,10,11,12,13,14,15);
#pragma unroll
        for (int j = 0; j < 4; ++j) {
            // B fragment: 32x16 f16, lane holds column (lr) = W row, K {0..15}/{16..31}
            const _Float16* bp = W + (size_t)(col0 + j * 16 + lr) * LDP + kk + (hi ? 16 : 0);
            h16 b = *(const h16*)bp;
            acc[j] = __builtin_amdgcn_wmma_f32_16x16x32_f16(
                false, a, false, b, (short)0, acc[j], false, false);
        }
    }

    const int rbase = row0 + (hi ? 8 : 0);
#pragma unroll
    for (int j = 0; j < 4; ++j) {
        int col = col0 + j * 16 + lr;
        if (col < C) {
            float bv = bias[col];
#pragma unroll
            for (int r = 0; r < 8; ++r)
                out[(size_t)(rbase + r) * LDP + col] = acc[j][r] + bv;
        }
    }
}

// ---------------------------------------------------------------- segment softmax passes
__global__ __launch_bounds__(256)
void k_init_seg(unsigned* __restrict__ segmax, float* __restrict__ segsum) {
    int i = blockIdx.x * 256 + threadIdx.x;
    if (i < N_NODES) { segmax[i] = f2ord(-INFINITY); segsum[i] = 0.f; }
}

__global__ __launch_bounds__(256)
void k_zero(float* __restrict__ p, int n) {
    int i = blockIdx.x * 256 + threadIdx.x;
    if (i < n) p[i] = 0.f;
}

// one wave per edge: logit = att . leaky_relu(xl[src] + xr[dst] + ea*We)
__global__ __launch_bounds__(256)
void k_logits(const float* __restrict__ xl, const float* __restrict__ xr,
              const float* __restrict__ ea, const float* __restrict__ We,
              const float* __restrict__ att, const int* __restrict__ src,
              const int* __restrict__ dst, float* __restrict__ logits,
              unsigned* __restrict__ segmax) {
    int wid = threadIdx.x >> 5, lane = threadIdx.x & 31;
    int e = blockIdx.x * 8 + wid;
    if (e >= N_EDGES) return;
    int s = src[e], d = dst[e];
    float eav = ea[e];
    const float* ps = xl + (size_t)s * LDP;
    const float* pd = xr + (size_t)d * LDP;
    float sum = 0.f;
    for (int c = lane; c < HID; c += 32) {
        float v = ps[c] + pd[c] + eav * We[c];
        v = (v > 0.f) ? v : NEG_SLOPE * v;
        sum += att[c] * v;
    }
#pragma unroll
    for (int off = 16; off; off >>= 1) sum += __shfl_xor(sum, off, 32);
    if (lane == 0) {
        logits[e] = sum;
        atomicMax(&segmax[d], f2ord(sum));
    }
}

__global__ __launch_bounds__(256)
void k_alpha(const float* __restrict__ logits, const int* __restrict__ dst,
             const unsigned* __restrict__ segmax, float* __restrict__ alpha,
             float* __restrict__ segsum) {
    int e = blockIdx.x * 256 + threadIdx.x;
    if (e >= N_EDGES) return;
    int d = dst[e];
    float a = __expf(logits[e] - ord2f(segmax[d]));
    alpha[e] = a;
    atomicAdd(&segsum[d], a);
}

// one block per edge: agg[dst] += xl[src] * alpha/denom
__global__ __launch_bounds__(256)
void k_scatter(const float* __restrict__ xl, const float* __restrict__ alpha,
               const float* __restrict__ segsum, const int* __restrict__ src,
               const int* __restrict__ dst, float* __restrict__ agg) {
    int e = blockIdx.x;
    int s = src[e], d = dst[e];
    float w = alpha[e] / (segsum[d] + 1e-16f);
    const float* xs = xl + (size_t)s * LDP;
    float* ad = agg + (size_t)d * LDP;
    for (int c = threadIdx.x; c < HID; c += 256)
        atomicAdd(&ad[c], xs[c] * w);
}

// ---------------------------------------------------------------- activations
// layer-1: h = relu(agg + bias), emitted directly as f16 with zero K-padding
__global__ __launch_bounds__(256)
void k_relu_to_f16(const float* __restrict__ agg, const float* __restrict__ bias,
                   _Float16* __restrict__ hf) {
    int i = blockIdx.x * 256 + threadIdx.x;
    if (i >= N_NODES * LDP) return;
    int c = i & (LDP - 1);
    float v = 0.f;
    if (c < HID) { v = agg[i] + bias[c]; v = v > 0.f ? v : 0.f; }
    hf[i] = (_Float16)v;
}

// layer-2: out = sigmoid(agg + bias), compact [N, N_CLS]
__global__ __launch_bounds__(256)
void k_sigmoid(const float* __restrict__ agg, const float* __restrict__ bias,
               float* __restrict__ out) {
    int i = blockIdx.x * 256 + threadIdx.x;
    if (i >= N_NODES * N_CLS) return;
    int n = i / N_CLS, c = i - n * N_CLS;
    float v = agg[(size_t)n * LDP + c] + bias[c];
    out[i] = 1.f / (1.f + __expf(-v));
}

// ---------------------------------------------------------------- launcher
extern "C" void kernel_launch(void* const* d_in, const int* in_sizes, int n_in,
                              void* d_out, int out_size, void* d_ws, size_t ws_size,
                              hipStream_t stream) {
    (void)in_sizes; (void)n_in; (void)out_size;
    const float* x    = (const float*)d_in[0];
    const int*   ei   = (const int*)d_in[1];
    const float* ea   = (const float*)d_in[2];
    const float* w1l  = (const float*)d_in[3];
    const float* b1l  = (const float*)d_in[4];
    const float* w1r  = (const float*)d_in[5];
    const float* b1r  = (const float*)d_in[6];
    const float* w1e  = (const float*)d_in[7];
    const float* att1 = (const float*)d_in[8];
    const float* bias1= (const float*)d_in[9];
    const float* w2l  = (const float*)d_in[10];
    const float* b2l  = (const float*)d_in[11];
    const float* w2r  = (const float*)d_in[12];
    const float* b2r  = (const float*)d_in[13];
    const float* w2e  = (const float*)d_in[14];
    const float* att2 = (const float*)d_in[15];
    const float* bias2= (const float*)d_in[16];
    const int* src = ei;
    const int* dst = ei + N_EDGES;

    // workspace layout
    char* ws = (char*)d_ws;
    size_t off = 0;
    auto take = [&](size_t bytes) { size_t p = off; off += (bytes + 255) & ~(size_t)255; return p; };
    size_t o_xf16 = take((size_t)N_NODES * LDP * 2);
    size_t o_wA   = take((size_t)LDP * LDP * 2);
    size_t o_wB   = take((size_t)LDP * LDP * 2);
    size_t o_xl   = take((size_t)N_NODES * LDP * 4);
    size_t o_xr   = take((size_t)N_NODES * LDP * 4);
    size_t o_agg  = take((size_t)N_NODES * LDP * 4);
    size_t o_log  = take((size_t)N_EDGES * 4);
    size_t o_alp  = take((size_t)N_EDGES * 4);
    size_t o_smx  = take((size_t)N_NODES * 4);
    size_t o_ssm  = take((size_t)N_NODES * 4);
    if (ws_size < off) return;  // insufficient scratch: bail rather than fault

    _Float16* xf16 = (_Float16*)(ws + o_xf16);
    _Float16* wA   = (_Float16*)(ws + o_wA);
    _Float16* wB   = (_Float16*)(ws + o_wB);
    float* xl      = (float*)(ws + o_xl);
    float* xr      = (float*)(ws + o_xr);
    float* agg     = (float*)(ws + o_agg);
    float* logits  = (float*)(ws + o_log);
    float* alpha   = (float*)(ws + o_alp);
    unsigned* smx  = (unsigned*)(ws + o_smx);
    float* ssm     = (float*)(ws + o_ssm);

    const int B = 256;
    const int gFeat  = (N_NODES * LDP + B - 1) / B;      // 80000
    const int gW     = (LDP * LDP) / B;                  // 4096
    const int gNode  = (N_NODES + B - 1) / B;
    const int gEdge  = (N_EDGES + B - 1) / B;
    const int gLog   = N_EDGES / 8;                      // 1 wave / edge
    const dim3 gGemm((MTILES + 7) / 8, LDP / 64);        // (157, 16)

    // ---------------- layer 1
    k_cvt_x<<<gFeat, B, 0, stream>>>(x, xf16, N_NODES * F_IN);
    k_cvt_w<<<gW, B, 0, stream>>>(w1l, wA, HID, F_IN);
    k_cvt_w<<<gW, B, 0, stream>>>(w1r, wB, HID, F_IN);
    k_gemm_bias<<<gGemm, B, 0, stream>>>(xf16, wA, b1l, xl, HID);
    k_gemm_bias<<<gGemm, B, 0, stream>>>(xf16, wB, b1r, xr, HID);
    k_init_seg<<<gNode, B, 0, stream>>>(smx, ssm);
    k_zero<<<gFeat, B, 0, stream>>>(agg, N_NODES * LDP);
    k_logits<<<gLog, B, 0, stream>>>(xl, xr, ea, w1e, att1, src, dst, logits, smx);
    k_alpha<<<gEdge, B, 0, stream>>>(logits, dst, smx, alpha, ssm);
    k_scatter<<<N_EDGES, B, 0, stream>>>(xl, alpha, ssm, src, dst, agg);
    k_relu_to_f16<<<gFeat, B, 0, stream>>>(agg, bias1, xf16);  // h (f16, padded) reuses xf16

    // ---------------- layer 2
    k_cvt_w<<<gW, B, 0, stream>>>(w2l, wA, N_CLS, HID);
    k_cvt_w<<<gW, B, 0, stream>>>(w2r, wB, N_CLS, HID);
    k_gemm_bias<<<gGemm, B, 0, stream>>>(xf16, wA, b2l, xl, N_CLS);
    k_gemm_bias<<<gGemm, B, 0, stream>>>(xf16, wB, b2r, xr, N_CLS);
    k_init_seg<<<gNode, B, 0, stream>>>(smx, ssm);
    k_zero<<<gFeat, B, 0, stream>>>(agg, N_NODES * LDP);
    k_logits<<<gLog, B, 0, stream>>>(xl, xr, ea, w2e, att2, src, dst, logits, smx);
    k_alpha<<<gEdge, B, 0, stream>>>(logits, dst, smx, alpha, ssm);
    k_scatter<<<N_EDGES, B, 0, stream>>>(xl, alpha, ssm, src, dst, agg);
    k_sigmoid<<<(N_NODES * N_CLS + B - 1) / B, B, 0, stream>>>(agg, bias2, (float*)d_out);
}
